// MaskedMHA_30322469110280
// MI455X (gfx1250) — matrix operand order repaired and checked by reading, not compiled, not run on hardware
//
#include <hip/hip_runtime.h>
#include <hip/hip_bf16.h>

typedef _Float16 f16;
typedef __attribute__((ext_vector_type(16))) _Float16 v16h;
typedef __attribute__((ext_vector_type(8)))  _Float16 v8h;
typedef __attribute__((ext_vector_type(8)))  float    v8f;
typedef __attribute__((ext_vector_type(4)))  float    v4f;

#define DEV __device__ __forceinline__

static constexpr int BATCH  = 2;
static constexpr int SEQ    = 4096;
static constexpr int DMODEL = 256;
static constexpr int NHEAD  = 8;
static constexpr int DKH    = 32;   // head dim == WMMA K-dim for f16
static constexpr int D3     = 768;
static constexpr int MROWS  = BATCH * SEQ;  // 8192
// softmax in base-2: (1/sqrt(32)) * log2(e)
static constexpr float SCALE_LOG2E = 0.25501817249737415f;
static constexpr float NEG_INF = -__builtin_inff();

// ---------- WMMA fragment helpers (wave32, ISA 7.12.2 layouts) ----------
DEV v16h load_a_frag(const f16* __restrict__ p, int stride, int lane) {
  const int m = lane & 15, hh = lane >> 4;
  const f16* row = p + (size_t)m * stride + hh * 8;
  v8h lo = *(const v8h*)(row);
  v8h hi = *(const v8h*)(row + 16);
  v16h r;
#pragma unroll
  for (int i = 0; i < 8; ++i) { r[i] = lo[i]; r[i + 8] = hi[i]; }
  return r;
}

DEV v16h load_bt_frag(const f16* __restrict__ p, int stride, int lane) {
  const int n = lane & 15, hh = lane >> 4;
  const f16* row = p + (size_t)n * stride + hh * 8;
  v8h lo = *(const v8h*)(row);
  v8h hi = *(const v8h*)(row + 16);
  v16h r;
#pragma unroll
  for (int i = 0; i < 8; ++i) { r[i] = lo[i]; r[i + 8] = hi[i]; }
  return r;
}

DEV v8f wmma_f16(v16h a, v16h b, v8f c) {
  return __builtin_amdgcn_wmma_f32_16x16x32_f16(false, a, false, b,
                                                (short)0, c, false, false);
}

// ---------- CDNA5 async global->LDS copy (ASYNCcnt-tracked, no VGPR data) ----
DEV void async_copy_b128(unsigned int lds_addr, const void* gaddr) {
  asm volatile("global_load_async_to_lds_b128 %0, %1, off"
               :: "v"(lds_addr), "v"(gaddr)
               : "memory");
}
DEV void wait_asynccnt0() {
  asm volatile("s_wait_asynccnt 0x0" ::: "memory");
}

// ---------- prep: W (K x N) f32  ->  W^T (N x K) f16 ----------
__global__ void transpose_to_f16(const float* __restrict__ W, f16* __restrict__ Wt,
                                 int K, int Nout) {
  int idx = blockIdx.x * blockDim.x + threadIdx.x;
  if (idx >= K * Nout) return;
  int nrow = idx / K;
  int kcol = idx - nrow * K;
  Wt[idx] = (f16)W[(size_t)kcol * Nout + nrow];
}

// ---------- prep: Vt[b][h][d][n] = qkv[b][n][512 + h*32 + d] ----------
__global__ void relayout_vt(const f16* __restrict__ qkv, f16* __restrict__ Vt) {
  int idx = blockIdx.x * blockDim.x + threadIdx.x;  // B*H*DKH*SEQ = 2,097,152
  int n    = idx & (SEQ - 1);
  int rest = idx >> 12;           // b*256 + h*32 + d
  int d = rest & 31;
  int h = (rest >> 5) & 7;
  int b = rest >> 8;
  Vt[idx] = qkv[((size_t)(b * SEQ + n)) * D3 + 2 * DMODEL + h * DKH + d];
}

// ---------- GEMM: Out(MxN) = A(MxK) @ Bt^T + bias ; Bt is (N x K) f16 ----------
template <bool A_IS_F32, bool OUT_F32>
__global__ void gemm_bias_kernel(const void* __restrict__ Ap,
                                 const f16* __restrict__ Bt,
                                 const float* __restrict__ bias,
                                 void* __restrict__ Outp,
                                 int M, int K, int Nout) {
  __shared__ alignas(16) f16 Als[64 * 32];
  __shared__ alignas(16) f16 Bls[64 * 32];
  const int lane = threadIdx.x & 31;
  const int w    = threadIdx.x >> 5;
  const int m0   = blockIdx.y * 64;
  const int n0   = blockIdx.x * 64;
  const int ar   = threadIdx.x >> 1;
  const int ac   = (threadIdx.x & 1) * 16;

  v8f acc[4] = {v8f{}, v8f{}, v8f{}, v8f{}};

  for (int kk = 0; kk < K; kk += 32) {
    if constexpr (A_IS_F32) {
      const float* src = (const float*)Ap + (size_t)(m0 + ar) * K + kk + ac;
      f16* dst = &Als[ar * 32 + ac];
#pragma unroll
      for (int i = 0; i < 16; i += 4) {
        v4f c = *(const v4f*)(src + i);
        dst[i + 0] = (f16)c[0]; dst[i + 1] = (f16)c[1];
        dst[i + 2] = (f16)c[2]; dst[i + 3] = (f16)c[3];
      }
    } else {
      const f16* src = (const f16*)Ap + (size_t)(m0 + ar) * K + kk + ac;
      *(v8h*)&Als[ar * 32 + ac]     = *(const v8h*)(src);
      *(v8h*)&Als[ar * 32 + ac + 8] = *(const v8h*)(src + 8);
    }
    {
      const f16* src = Bt + (size_t)(n0 + ar) * K + kk + ac;
      *(v8h*)&Bls[ar * 32 + ac]     = *(const v8h*)(src);
      *(v8h*)&Bls[ar * 32 + ac + 8] = *(const v8h*)(src + 8);
    }
    __syncthreads();

    v16h a = load_a_frag(&Als[(w * 16) * 32], 32, lane);
#pragma unroll
    for (int j = 0; j < 4; ++j) {
      v16h b = load_bt_frag(&Bls[(j * 16) * 32], 32, lane);
      acc[j] = wmma_f16(a, b, acc[j]);
    }
    __syncthreads();
  }

  const int nn = lane & 15, hh = lane >> 4;
#pragma unroll
  for (int j = 0; j < 4; ++j) {
    const int ng = n0 + j * 16 + nn;
    const float bv = bias[ng];
#pragma unroll
    for (int v = 0; v < 8; ++v) {
      const int mg = m0 + w * 16 + v + 8 * hh;
      const float r = acc[j][v] + bv;
      if constexpr (OUT_F32) ((float*)Outp)[(size_t)mg * Nout + ng] = r;
      else                   ((f16*)Outp)[(size_t)mg * Nout + ng] = (f16)r;
    }
  }
}

// ---------- flash attention with key mask ----------
// grid: (SEQ/64, NHEAD, BATCH); block 128 (4 waves, 16 q-rows each), 64-key tiles.
__global__ void attn_kernel(const f16* __restrict__ qkv,
                            const f16* __restrict__ Vt,
                            const unsigned char* __restrict__ mask,
                            f16* __restrict__ Obuf) {
  __shared__ alignas(16) f16 Kls[64 * 32];      // keys x dk  (== B^T for QK^T)
  __shared__ alignas(16) f16 Vtls[32 * 64];     // dk x keys  (== B^T for PV)
  __shared__ alignas(16) f16 Pls[4][16 * 64];   // per-wave prob tile (D->A relayout)

  const int w    = threadIdx.x >> 5;
  const int lane = threadIdx.x & 31;
  const int nn   = lane & 15, hh = lane >> 4;
  const int b    = blockIdx.z;
  const int h    = blockIdx.y;
  const int q0   = blockIdx.x * 64;

  const unsigned int kls_base  = (unsigned int)(size_t)&Kls[0];
  const unsigned int vtls_base = (unsigned int)(size_t)&Vtls[0];

  // Q fragment: resident for the whole key loop
  const f16* qbase = qkv + ((size_t)(b * SEQ + q0 + w * 16)) * D3 + h * DKH;
  const v16h qa = load_a_frag(qbase, D3, lane);

  const f16* kgbase  = qkv + (size_t)b * SEQ * D3 + DMODEL + h * DKH;  // + key*D3
  const f16* vtgbase = Vt + ((size_t)(b * NHEAD + h) * DKH) * SEQ;     // + d*SEQ
  const unsigned char* mb = mask + (size_t)b * SEQ;

  v8f o0 = {}, o1 = {};
  float mrow[8], lrow[8];
#pragma unroll
  for (int v = 0; v < 8; ++v) { mrow[v] = NEG_INF; lrow[v] = 0.f; }

  for (int k0 = 0; k0 < SEQ; k0 += 64) {
    // async-stage K (64x32) and Vt (32x64) tiles: 2 x 16B per matrix per thread
#pragma unroll
    for (int r = 0; r < 2; ++r) {
      const int s = threadIdx.x + r * 128;          // 0..255
      const int key = s >> 2, kp = s & 3;           // K tile: 4 x 16B per key row
      async_copy_b128(kls_base + key * 64 + kp * 16,
                      kgbase + (size_t)(k0 + key) * D3 + kp * 8);
      const int d = s >> 3, vp = s & 7;             // Vt tile: 8 x 16B per d row
      async_copy_b128(vtls_base + d * 128 + vp * 16,
                      vtgbase + (size_t)d * SEQ + k0 + vp * 8);
    }
    wait_asynccnt0();
    __syncthreads();

    // S = Q @ K^T : four 16-key sub-tiles, one WMMA each (K-dim = dk = 32)
    v8f z = {};
    v8f s4[4];
#pragma unroll
    for (int j = 0; j < 4; ++j) {
      v16h kb = load_bt_frag(&Kls[(j * 16) * 32], 32, lane);
      s4[j] = wmma_f16(qa, kb, z);
    }

    bool msk[4];
#pragma unroll
    for (int j = 0; j < 4; ++j) msk[j] = mb[k0 + j * 16 + nn] != 0;

    // online softmax in base-2; row = v + 8*hh spread across a 16-lane half
#pragma unroll
    for (int v = 0; v < 8; ++v) {
      float aj[4];
#pragma unroll
      for (int j = 0; j < 4; ++j)
        aj[j] = msk[j] ? NEG_INF : s4[j][v] * SCALE_LOG2E;
      float mx = fmaxf(fmaxf(aj[0], aj[1]), fmaxf(aj[2], aj[3]));
      mx = fmaxf(mx, __shfl_xor(mx, 1, 32));
      mx = fmaxf(mx, __shfl_xor(mx, 2, 32));
      mx = fmaxf(mx, __shfl_xor(mx, 4, 32));
      mx = fmaxf(mx, __shfl_xor(mx, 8, 32));
      const float mnew = fmaxf(mrow[v], mx);
      float alpha, p[4], rs;
      if (mnew == NEG_INF) {
        alpha = 1.f; p[0] = p[1] = p[2] = p[3] = 0.f; rs = 0.f;
      } else {
        alpha = exp2f(mrow[v] - mnew);
#pragma unroll
        for (int j = 0; j < 4; ++j) p[j] = exp2f(aj[j] - mnew);
        rs = (p[0] + p[1]) + (p[2] + p[3]);
      }
      rs += __shfl_xor(rs, 1, 32);
      rs += __shfl_xor(rs, 2, 32);
      rs += __shfl_xor(rs, 4, 32);
      rs += __shfl_xor(rs, 8, 32);
      lrow[v] = lrow[v] * alpha + rs;
      mrow[v] = mnew;
      o0[v] *= alpha;
      o1[v] *= alpha;
      f16* pr = &Pls[w][(v + 8 * hh) * 64];
      pr[nn]      = (f16)p[0];
      pr[16 + nn] = (f16)p[1];
      pr[32 + nn] = (f16)p[2];
      pr[48 + nn] = (f16)p[3];   // same-wave DS ops are in-order
    }

    // O += P @ V : P is 16x64 (two A frags); V^T tile gives both dk halves
    v16h pa0 = load_a_frag(&Pls[w][0],  64, lane);
    v16h pa1 = load_a_frag(&Pls[w][32], 64, lane);
    o0 = wmma_f16(pa0, load_bt_frag(&Vtls[0],            64, lane), o0);
    o0 = wmma_f16(pa1, load_bt_frag(&Vtls[32],           64, lane), o0);
    o1 = wmma_f16(pa0, load_bt_frag(&Vtls[16 * 64],      64, lane), o1);
    o1 = wmma_f16(pa1, load_bt_frag(&Vtls[16 * 64 + 32], 64, lane), o1);
    __syncthreads();
  }

  // normalize and write f16 O in (B,N,256) layout for the output projection
#pragma unroll
  for (int v = 0; v < 8; ++v) {
    const int mg = q0 + w * 16 + v + 8 * hh;
    const float inv = 1.0f / lrow[v];
    f16* orow = Obuf + ((size_t)(b * SEQ + mg)) * DMODEL + h * DKH;
    orow[nn]      = (f16)(o0[v] * inv);
    orow[16 + nn] = (f16)(o1[v] * inv);
  }
}

extern "C" void kernel_launch(void* const* d_in, const int* in_sizes, int n_in,
                              void* d_out, int out_size, void* d_ws, size_t ws_size,
                              hipStream_t stream) {
  (void)in_sizes; (void)n_in; (void)out_size; (void)ws_size;
  const float*         x     = (const float*)d_in[0];
  const unsigned char* mask  = (const unsigned char*)d_in[1];  // numpy bool8
  const float*         Wqkv  = (const float*)d_in[2];
  const float*         bqkv  = (const float*)d_in[3];
  const float*         Wproj = (const float*)d_in[4];
  const float*         bproj = (const float*)d_in[5];
  float* out = (float*)d_out;

  // workspace: qkv f16 | O f16 | V^T f16 | Wqkv^T f16 | Wproj^T f16  (~21.5 MB)
  char* ws = (char*)d_ws;
  f16* qkv  = (f16*)ws;
  f16* Obuf = (f16*)(ws + (size_t)MROWS * D3 * 2);
  f16* VtB  = (f16*)(ws + (size_t)MROWS * D3 * 2 + (size_t)MROWS * DMODEL * 2);
  f16* WtQ  = VtB + (size_t)BATCH * NHEAD * DKH * SEQ;
  f16* WtP  = WtQ + (size_t)D3 * DMODEL;

  transpose_to_f16<<<(D3 * DMODEL + 255) / 256, 256, 0, stream>>>(Wqkv, WtQ, DMODEL, D3);
  transpose_to_f16<<<(DMODEL * DMODEL + 255) / 256, 256, 0, stream>>>(Wproj, WtP, DMODEL, DMODEL);

  gemm_bias_kernel<true, false>
      <<<dim3(D3 / 64, MROWS / 64), 128, 0, stream>>>(x, WtQ, bqkv, qkv,
                                                      MROWS, DMODEL, D3);

  relayout_vt<<<(BATCH * NHEAD * DKH * SEQ) / 256, 256, 0, stream>>>(qkv, VtB);

  attn_kernel<<<dim3(SEQ / 64, NHEAD, BATCH), 128, 0, stream>>>(qkv, VtB, mask, Obuf);

  gemm_bias_kernel<false, true>
      <<<dim3(DMODEL / 64, MROWS / 64), 128, 0, stream>>>(Obuf, WtP, bproj, out,
                                                          MROWS, DMODEL, DMODEL);
}
